// RNN_34351148433726
// MI455X (gfx1250) — matrix operand order repaired and checked by reading, not compiled, not run on hardware
//
#include <hip/hip_runtime.h>
#include <hip/hip_bf16.h>

typedef __attribute__((ext_vector_type(16))) _Float16 v16h;
typedef __attribute__((ext_vector_type(8)))  _Float16 v8h;
typedef __attribute__((ext_vector_type(8)))  float    v8f;
typedef __attribute__((ext_vector_type(4)))  int      v4i;

#define BATCH   8192
#define SEQ     28
#define INPUT   28
#define HID     128
#define GATES   512   // 4*HID
#define TILE_B  64
#define NT      512   // threads per block = 16 waves
#define NCLS    10

// ---- CDNA5 hardware tanh (V_TANH_F32) ----
#if __has_builtin(__builtin_amdgcn_tanhf)
  #define TANH_HW(x) __builtin_amdgcn_tanhf(x)
#elif __has_builtin(__builtin_amdgcn_tanh_f32)
  #define TANH_HW(x) __builtin_amdgcn_tanh_f32(x)
#else
  #define TANH_HW(x) (2.0f / (1.0f + __expf(-2.0f * (x))) - 1.0f)
#endif
__device__ __forceinline__ float tanh_hw(float x) { return TANH_HW(x); }
__device__ __forceinline__ float sigf(float x)    { return 0.5f * tanh_hw(0.5f * x) + 0.5f; }

// ---- CDNA5 async global->LDS copy (ASYNCcnt) ----
#if __has_builtin(__builtin_amdgcn_global_load_async_to_lds_b128)
  #define HAS_ASYNC_LDS 1
#endif

#ifdef HAS_ASYNC_LDS
typedef __attribute__((address_space(1))) v4i* as1_v4i;
typedef __attribute__((address_space(3))) v4i* as3_v4i;
__device__ __forceinline__ void async_b128(const void* g, void* l) {
  __builtin_amdgcn_global_load_async_to_lds_b128(
      (as1_v4i)(uintptr_t)g,
      (as3_v4i)(uint32_t)(uintptr_t)l,
      0, 0);
}
__device__ __forceinline__ void wait_async0() {
#if __has_builtin(__builtin_amdgcn_s_wait_asynccnt)
  __builtin_amdgcn_s_wait_asynccnt(0);
#else
  asm volatile("s_wait_asynccnt 0x0" ::: "memory");
#endif
}
#endif

// A fragment (16x32 f16): lanes 0-15 row M=l16 hold K {0..15}; lanes 16-31 hold K {16..31}
__device__ __forceinline__ v16h ldsA(const _Float16* rowbase, int koff, int hi) {
  v8h c0 = *(const v8h*)(rowbase + koff + hi * 16);
  v8h c1 = *(const v8h*)(rowbase + koff + hi * 16 + 8);
  return __builtin_shufflevector(c0, c1, 0,1,2,3,4,5,6,7,8,9,10,11,12,13,14,15);
}
// B fragment (32x16 f16): lane holds one N column, 16 contiguous K halfs
__device__ __forceinline__ v16h ldsB(const _Float16* p) {
  v8h c0 = *(const v8h*)(p);
  v8h c1 = *(const v8h*)(p + 8);
  return __builtin_shufflevector(c0, c1, 0,1,2,3,4,5,6,7,8,9,10,11,12,13,14,15);
}
__device__ __forceinline__ v8f wmma16(v16h a, v16h b, v8f c) {
  return __builtin_amdgcn_wmma_f32_16x16x32_f16(false, a, false, b, (short)0, c, false, false);
}

// Wave -> tile mapping (16 waves): wave = hcol(0..7) + 8*rp(0..1).
// Each wave owns 2 row-tiles {2rp, 2rp+1} x 1 hidden col-tile hcol.
// All 16 W_hh B-fragments (4 kb x 4 gates) live in registers across the time loop;
// each is reused by both row-tiles. acc = 2x4 v8f, cst = 2 v8f. W_ih fragments come
// from LDS (each feeds 2 WMMAs) to stay under the 256-VGPR cap at 16 waves/WGP.

// ---------------- Layer 0: (B,T,28) fp32 -> h1 (T,B,128) f16 ----------------
__global__ __launch_bounds__(NT) void lstm_layer0(
    const float* __restrict__ x,   const float* __restrict__ Wih,
    const float* __restrict__ Whh, const float* __restrict__ bih,
    const float* __restrict__ bhh, _Float16* __restrict__ h1out)
{
  __shared__ __align__(16) _Float16 sWhh[GATES * HID];   // 128 KB (staging for reg fragments)
  __shared__ __align__(16) _Float16 sWih[GATES * 32];    //  32 KB (K padded 28->32)
  __shared__ __align__(16) _Float16 sH  [TILE_B * HID];  //  16 KB
  __shared__ __align__(16) _Float16 sX  [TILE_B * 32];   //   4 KB
  __shared__ float sBias[GATES];                         //   2 KB

  const int tid = threadIdx.x;
  const int b0  = blockIdx.x * TILE_B;

  for (int i = tid; i < GATES * HID; i += NT) sWhh[i] = (_Float16)Whh[i];
  for (int i = tid; i < GATES * 32; i += NT) {
    int r = i >> 5, c = i & 31;
    sWih[i] = (c < INPUT) ? (_Float16)Wih[r * INPUT + c] : (_Float16)0.0f;
  }
  for (int i = tid; i < GATES; i += NT) sBias[i] = bih[i] + bhh[i];
  for (int i = tid; i < TILE_B * HID; i += NT) sH[i] = (_Float16)0.0f;

  const int wave = tid >> 5, lane = tid & 31;
  const int l16 = lane & 15, hi = lane >> 4;
  const int hcol = wave & 7;              // hidden col-tile (0..7)
  const int rp   = wave >> 3;             // row-tile pair (0..1)
  const int mrow0 = (rp * 2 + 0) * 16 + l16;
  const int mrow1 = (rp * 2 + 1) * 16 + l16;

  __syncthreads();

  // Hoist loop-invariant W_hh fragments + bias into registers
  float bsum[4];
  v16h whhf[4][4];   // [kb][gate]
  #pragma unroll
  for (int g = 0; g < 4; ++g) {
    const int grow = g * HID + hcol * 16 + l16;
    bsum[g] = sBias[grow];
    #pragma unroll
    for (int kb = 0; kb < 4; ++kb)
      whhf[kb][g] = ldsB(&sWhh[grow * HID + kb * 32 + hi * 16]);
  }

  const v8f vzero = {};
  v8f cst[2];                              // cell state, D-matrix register layout
  #pragma unroll
  for (int r = 0; r < 2; ++r) cst[r] = vzero;

  // software-pipelined x staging
  float xv[4];
  #pragma unroll
  for (int k = 0; k < 4; ++k) {
    int i = tid + k * NT, r = i >> 5, c = i & 31;
    xv[k] = (c < INPUT) ? x[((size_t)(b0 + r) * SEQ + 0) * INPUT + c] : 0.0f;
  }

  for (int t = 0; t < SEQ; ++t) {
    #pragma unroll
    for (int k = 0; k < 4; ++k) { int i = tid + k * NT; sX[i] = (_Float16)xv[k]; }
    __syncthreads();

    if (t + 1 < SEQ) {
      #pragma unroll
      for (int k = 0; k < 4; ++k) {
        int i = tid + k * NT, r = i >> 5, c = i & 31;
        xv[k] = (c < INPUT) ? x[((size_t)(b0 + r) * SEQ + (t + 1)) * INPUT + c] : 0.0f;
      }
    }

    v8f acc[2][4];
    { // input projection, K = 32 (padded): each W_ih fragment feeds 2 WMMAs
      v16h ax0 = ldsA(&sX[mrow0 * 32], 0, hi);
      v16h ax1 = ldsA(&sX[mrow1 * 32], 0, hi);
      #pragma unroll
      for (int g = 0; g < 4; ++g) {
        const int grow = g * HID + hcol * 16 + l16;
        v16h b = ldsB(&sWih[grow * 32 + hi * 16]);
        acc[0][g] = wmma16(ax0, b, vzero);
        acc[1][g] = wmma16(ax1, b, vzero);
      }
    }
    #pragma unroll
    for (int kb = 0; kb < 4; ++kb) { // recurrence: B operands all in registers
      v16h ah0 = ldsA(&sH[mrow0 * HID], kb * 32, hi);
      v16h ah1 = ldsA(&sH[mrow1 * HID], kb * 32, hi);
      #pragma unroll
      for (int g = 0; g < 4; ++g) {
        acc[0][g] = wmma16(ah0, whhf[kb][g], acc[0][g]);
        acc[1][g] = wmma16(ah1, whhf[kb][g], acc[1][g]);
      }
    }
    __syncthreads();  // all reads of sH (h_{t-1}) complete

    #pragma unroll
    for (int r = 0; r < 2; ++r) {
      #pragma unroll
      for (int e = 0; e < 8; ++e) {
        float ig = sigf   (acc[r][0][e] + bsum[0]);
        float fg = sigf   (acc[r][1][e] + bsum[1]);
        float gg = tanh_hw(acc[r][2][e] + bsum[2]);
        float og = sigf   (acc[r][3][e] + bsum[3]);
        float c  = fg * cst[r][e] + ig * gg;
        cst[r][e] = c;
        float h  = og * tanh_hw(c);
        const int row = (rp * 2 + r) * 16 + e + hi * 8;   // f32 C/D layout
        sH[row * HID + hcol * 16 + l16] = (_Float16)h;
      }
    }
    __syncthreads();

    { // stream h_t tile to global (f16); sH is not rewritten until 2 barriers later
      _Float16* dst = h1out + ((size_t)t * BATCH + b0) * HID;
      #pragma unroll
      for (int k = 0; k < 2; ++k) {
        int ch = tid + k * NT, row = ch >> 4, part = ch & 15;
        *(v8h*)(dst + row * HID + part * 8) = *(const v8h*)(&sH[row * HID + part * 8]);
      }
    }
  }
}

// ------------- Layer 1 + FC + log_softmax: h1 (T,B,128) f16 -> out (B,10) f32 -------------
__global__ __launch_bounds__(NT) void lstm_layer1_fc(
    const _Float16* __restrict__ h1, const float* __restrict__ Wih,
    const float* __restrict__ Whh,   const float* __restrict__ bih,
    const float* __restrict__ bhh,   const float* __restrict__ Wfc,
    const float* __restrict__ bfc,   float* __restrict__ out)
{
  __shared__ __align__(16) _Float16 sWih[GATES * HID];     // 128 KB
  __shared__ __align__(16) _Float16 sWhh[GATES * HID];     // 128 KB (staging for reg fragments)
#ifdef HAS_ASYNC_LDS
  __shared__ __align__(16) _Float16 sHin[2][TILE_B * HID]; //  32 KB (double buffer)
#else
  __shared__ __align__(16) _Float16 sHin[1][TILE_B * HID]; //  16 KB
#endif
  __shared__ __align__(16) _Float16 sH  [TILE_B * HID];    //  16 KB
  __shared__ float sBias[GATES];
  __shared__ float sLogit[TILE_B * NCLS];

  const int tid = threadIdx.x;
  const int b0  = blockIdx.x * TILE_B;

  for (int i = tid; i < GATES * HID; i += NT) sWih[i] = (_Float16)Wih[i];
  for (int i = tid; i < GATES * HID; i += NT) sWhh[i] = (_Float16)Whh[i];
  for (int i = tid; i < GATES; i += NT) sBias[i] = bih[i] + bhh[i];
  for (int i = tid; i < TILE_B * HID; i += NT) sH[i] = (_Float16)0.0f;

  const int wave = tid >> 5, lane = tid & 31;
  const int l16 = lane & 15, hi = lane >> 4;
  const int hcol = wave & 7;
  const int rp   = wave >> 3;
  const int mrow0 = (rp * 2 + 0) * 16 + l16;
  const int mrow1 = (rp * 2 + 1) * 16 + l16;

  __syncthreads();

  // Hoist W_hh fragments + bias; W_ih stays in LDS (each fragment feeds 2 WMMAs)
  float bsum[4];
  v16h whhf[4][4];
  #pragma unroll
  for (int g = 0; g < 4; ++g) {
    const int grow = g * HID + hcol * 16 + l16;
    bsum[g] = sBias[grow];
    #pragma unroll
    for (int kb = 0; kb < 4; ++kb)
      whhf[kb][g] = ldsB(&sWhh[grow * HID + kb * 32 + hi * 16]);
  }

  const v8f vzero = {};
  v8f cst[2];
  #pragma unroll
  for (int r = 0; r < 2; ++r) cst[r] = vzero;

#ifdef HAS_ASYNC_LDS
  { // kick off async copy of tile 0
    const _Float16* src = h1 + ((size_t)0 * BATCH + b0) * HID;
    #pragma unroll
    for (int k = 0; k < 2; ++k) {
      int ch = tid + k * NT, row = ch >> 4, part = ch & 15;
      async_b128(src + row * HID + part * 8, &sHin[0][row * HID + part * 8]);
    }
  }
#else
  v8h hv[2];
  #pragma unroll
  for (int k = 0; k < 2; ++k) {
    int ch = tid + k * NT, row = ch >> 4, part = ch & 15;
    hv[k] = *(const v8h*)(h1 + ((size_t)0 * BATCH + b0 + row) * HID + part * 8);
  }
#endif

  for (int t = 0; t < SEQ; ++t) {
#ifdef HAS_ASYNC_LDS
    const int cur = t & 1;
    wait_async0();
    __syncthreads();                      // sHin[cur] ready for everyone
    if (t + 1 < SEQ) {                    // overlap next tile fetch with WMMA phase
      const _Float16* src = h1 + ((size_t)(t + 1) * BATCH + b0) * HID;
      #pragma unroll
      for (int k = 0; k < 2; ++k) {
        int ch = tid + k * NT, row = ch >> 4, part = ch & 15;
        async_b128(src + row * HID + part * 8, &sHin[1 - cur][row * HID + part * 8]);
      }
    }
    const _Float16* sIn = sHin[cur];
#else
    const int cur = 0;
    #pragma unroll
    for (int k = 0; k < 2; ++k) {
      int ch = tid + k * NT, row = ch >> 4, part = ch & 15;
      *(v8h*)(&sHin[0][row * HID + part * 8]) = hv[k];
    }
    __syncthreads();
    if (t + 1 < SEQ) {
      #pragma unroll
      for (int k = 0; k < 2; ++k) {
        int ch = tid + k * NT, row = ch >> 4, part = ch & 15;
        hv[k] = *(const v8h*)(h1 + ((size_t)(t + 1) * BATCH + b0 + row) * HID + part * 8);
      }
    }
    const _Float16* sIn = sHin[cur];
#endif

    v8f acc[2][4];
    #pragma unroll
    for (int kb = 0; kb < 4; ++kb) { // input projection, K = 128; each B frag feeds 2 WMMAs
      v16h ax0 = ldsA(&sIn[mrow0 * HID], kb * 32, hi);
      v16h ax1 = ldsA(&sIn[mrow1 * HID], kb * 32, hi);
      #pragma unroll
      for (int g = 0; g < 4; ++g) {
        const int grow = g * HID + hcol * 16 + l16;
        v16h b = ldsB(&sWih[grow * HID + kb * 32 + hi * 16]);
        if (kb == 0) {
          acc[0][g] = wmma16(ax0, b, vzero);
          acc[1][g] = wmma16(ax1, b, vzero);
        } else {
          acc[0][g] = wmma16(ax0, b, acc[0][g]);
          acc[1][g] = wmma16(ax1, b, acc[1][g]);
        }
      }
    }
    #pragma unroll
    for (int kb = 0; kb < 4; ++kb) { // recurrence: B operands all in registers
      v16h ah0 = ldsA(&sH[mrow0 * HID], kb * 32, hi);
      v16h ah1 = ldsA(&sH[mrow1 * HID], kb * 32, hi);
      #pragma unroll
      for (int g = 0; g < 4; ++g) {
        acc[0][g] = wmma16(ah0, whhf[kb][g], acc[0][g]);
        acc[1][g] = wmma16(ah1, whhf[kb][g], acc[1][g]);
      }
    }
    __syncthreads();

    #pragma unroll
    for (int r = 0; r < 2; ++r) {
      #pragma unroll
      for (int e = 0; e < 8; ++e) {
        float ig = sigf   (acc[r][0][e] + bsum[0]);
        float fg = sigf   (acc[r][1][e] + bsum[1]);
        float gg = tanh_hw(acc[r][2][e] + bsum[2]);
        float og = sigf   (acc[r][3][e] + bsum[3]);
        float c  = fg * cst[r][e] + ig * gg;
        cst[r][e] = c;
        float h  = og * tanh_hw(c);
        const int row = (rp * 2 + r) * 16 + e + hi * 8;
        sH[row * HID + hcol * 16 + l16] = (_Float16)h;
      }
    }
    __syncthreads();
  }

  // FC: logits = h_T @ Wfc^T + bfc
  for (int idx = tid; idx < TILE_B * NCLS; idx += NT) {
    int row = idx / NCLS, c = idx % NCLS;
    float s = bfc[c];
    #pragma unroll 8
    for (int j = 0; j < HID; ++j)
      s += (float)sH[row * HID + j] * Wfc[c * HID + j];
    sLogit[idx] = s;
  }
  __syncthreads();

  if (tid < TILE_B) {
    float m = -1e30f;
    #pragma unroll
    for (int c = 0; c < NCLS; ++c) m = fmaxf(m, sLogit[tid * NCLS + c]);
    float s = 0.0f;
    #pragma unroll
    for (int c = 0; c < NCLS; ++c) s += __expf(sLogit[tid * NCLS + c] - m);
    float lse = m + __logf(s);
    #pragma unroll
    for (int c = 0; c < NCLS; ++c)
      out[(size_t)(b0 + tid) * NCLS + c] = sLogit[tid * NCLS + c] - lse;
  }
}

extern "C" void kernel_launch(void* const* d_in, const int* in_sizes, int n_in,
                              void* d_out, int out_size, void* d_ws, size_t ws_size,
                              hipStream_t stream) {
  const float* x     = (const float*)d_in[0];
  const float* W_ih0 = (const float*)d_in[1];
  const float* W_hh0 = (const float*)d_in[2];
  const float* b_ih0 = (const float*)d_in[3];
  const float* b_hh0 = (const float*)d_in[4];
  const float* W_ih1 = (const float*)d_in[5];
  const float* W_hh1 = (const float*)d_in[6];
  const float* b_ih1 = (const float*)d_in[7];
  const float* b_hh1 = (const float*)d_in[8];
  const float* W_fc  = (const float*)d_in[9];
  const float* b_fc  = (const float*)d_in[10];

  _Float16* h1 = (_Float16*)d_ws;   // needs SEQ*BATCH*HID*2 = 58.7 MB scratch

  dim3 grid(BATCH / TILE_B), block(NT);
  lstm_layer0<<<grid, block, 0, stream>>>(x, W_ih0, W_hh0, b_ih0, b_hh0, h1);
  lstm_layer1_fc<<<grid, block, 0, stream>>>(h1, W_ih1, W_hh1, b_ih1, b_hh1,
                                             W_fc, b_fc, (float*)d_out);
}
